// FusedAttention_3023656976929
// MI455X (gfx1250) — compile-verified, matrix operand order checked by reading, not run on hardware
//
#include <hip/hip_runtime.h>
#include <hip/hip_bf16.h>

// Problem constants (match reference)
#define Bb   4
#define Ss   2048
#define Hh   2048
#define NHh  16
#define HDd  128
#define N3   6144   // 3*H

#define AS1 __attribute__((address_space(1)))   // global
#define AS3 __attribute__((address_space(3)))   // LDS

typedef __attribute__((ext_vector_type(16))) __bf16 bf16x16;
typedef __attribute__((ext_vector_type(8)))  __bf16 bf16x8;
typedef __attribute__((ext_vector_type(8)))  float  f32x8;
typedef __attribute__((ext_vector_type(4)))  float  f32x4;
typedef int v4i_vs __attribute__((vector_size(16)));   // matches async builtin param

// ---------------------------------------------------------------------------
// CDNA5 async global->LDS copy (ASYNCcnt-tracked), with sync fallback.
// ---------------------------------------------------------------------------
#if __has_builtin(__builtin_amdgcn_global_load_async_to_lds_b128)
#define HAVE_ASYNC_LDS 1
#else
#define HAVE_ASYNC_LDS 0
#endif

static __device__ __forceinline__ void cp16_g2l(const AS1 __bf16* g, AS3 __bf16* l) {
#if HAVE_ASYNC_LDS
  __builtin_amdgcn_global_load_async_to_lds_b128(
      (AS1 v4i_vs*)g, (AS3 v4i_vs*)l, 0, 0);
#else
  *(AS3 bf16x8*)l = *(const AS1 bf16x8*)g;
#endif
}

static __device__ __forceinline__ void async_wait0() {
#if HAVE_ASYNC_LDS
#if __has_builtin(__builtin_amdgcn_s_wait_asynccnt)
  __builtin_amdgcn_s_wait_asynccnt(0);
#else
  asm volatile("s_wait_asynccnt 0x0" ::: "memory");
#endif
#endif
}

static __device__ __forceinline__ f32x8 f32x8_zero() {
  f32x8 z;
#pragma unroll
  for (int i = 0; i < 8; ++i) z[i] = 0.0f;
  return z;
}

// 16x32 bf16 WMMA A/B fragment loads (CDNA5 VGPR layout) from row-major
// storage, leading dimension `ld` elements. Two 16B loads per lane.
// Explicit address spaces: _l -> ds_load_b128, _g -> global_load_b128.
static __device__ __forceinline__ bf16x16 frag_combine(bf16x8 lo, bf16x8 hi) {
  bf16x16 out;
#pragma unroll
  for (int i = 0; i < 8; ++i) { out[i] = lo[i]; out[i + 8] = hi[i]; }
  return out;
}

static __device__ __forceinline__ bf16x16 load_frag_l(const AS3 __bf16* base, int ld) {
  const int lane = threadIdx.x & 31;
  const AS3 __bf16* p = base + (lane & 15) * ld + (lane >> 4) * 8;
  return frag_combine(*(const AS3 bf16x8*)p, *(const AS3 bf16x8*)(p + 16));
}

static __device__ __forceinline__ bf16x16 load_frag_g(const AS1 __bf16* base, int ld) {
  const int lane = threadIdx.x & 31;
  const AS1 __bf16* p = base + (lane & 15) * ld + (lane >> 4) * 8;
  return frag_combine(*(const AS1 bf16x8*)p, *(const AS1 bf16x8*)(p + 16));
}

static __device__ __forceinline__ f32x8 wmma_bf16(bf16x16 a, bf16x16 b, f32x8 c) {
  return __builtin_amdgcn_wmma_f32_16x16x32_bf16(false, a, false, b,
                                                 (short)0, c, false, false);
}

// ---------------------------------------------------------------------------
// fp32 [K][N] -> bf16 [N][K] tiled transpose-convert (one-time, weights).
// ---------------------------------------------------------------------------
__global__ __launch_bounds__(256) void cvt_transpose_bf16(const float* __restrict__ in,
                                                          __bf16* __restrict__ out,
                                                          int K, int N) {
  __shared__ __align__(16) __bf16 tile[64][72];   // 72: 16B-aligned rows, skewed
  const int t = threadIdx.x;
  const int nb = blockIdx.x * 64;
  const int kb = blockIdx.y * 64;
#pragma unroll
  for (int i = 0; i < 4; ++i) {
    int idx = i * 256 + t;            // 1024 quads
    int r  = idx >> 4;                // k row
    int c4 = (idx & 15) << 2;         // n col
    f32x4 v = *(const AS1 f32x4*)((const AS1 float*)in + (size_t)(kb + r) * N + nb + c4);
#pragma unroll
    for (int j = 0; j < 4; ++j) tile[c4 + j][r] = (__bf16)v[j];
  }
  __syncthreads();
#pragma unroll
  for (int i = 0; i < 2; ++i) {
    int idx = i * 256 + t;            // 512 octets
    int r  = idx >> 3;                // n row
    int c8 = (idx & 7) << 3;          // k col
    *(AS1 bf16x8*)((AS1 __bf16*)out + (size_t)(nb + r) * K + kb + c8) =
        *(const bf16x8*)(&tile[r][c8]);
  }
}

// ---------------------------------------------------------------------------
// Fused LayerNorm + bf16 cast. One row (H=2048) per workgroup, 256 threads.
// ---------------------------------------------------------------------------
__global__ __launch_bounds__(256) void ln_bf16(const float* __restrict__ x,
                                               const float* __restrict__ w,
                                               const float* __restrict__ bta,
                                               __bf16* __restrict__ xn) {
  const int row = blockIdx.x;
  const int t = threadIdx.x;
  const int lane = t & 31, wv = t >> 5;
  const AS1 float* xr = (const AS1 float*)x + (size_t)row * Hh;

  f32x4 v0 = *(const AS1 f32x4*)(xr + t * 8);
  f32x4 v1 = *(const AS1 f32x4*)(xr + t * 8 + 4);
  float s = 0.f, s2 = 0.f;
#pragma unroll
  for (int j = 0; j < 4; ++j) { s += v0[j] + v1[j]; s2 += v0[j]*v0[j] + v1[j]*v1[j]; }
#pragma unroll
  for (int m = 16; m >= 1; m >>= 1) {
    s  += __shfl_xor(s,  m, 32);
    s2 += __shfl_xor(s2, m, 32);
  }
  __shared__ float aS[8], aS2[8];
  if (lane == 0) { aS[wv] = s; aS2[wv] = s2; }
  __syncthreads();
  float ts = 0.f, ts2 = 0.f;
#pragma unroll
  for (int i = 0; i < 8; ++i) { ts += aS[i]; ts2 += aS2[i]; }
  const float mu   = ts * (1.0f / Hh);
  const float var  = ts2 * (1.0f / Hh) - mu * mu;
  const float rstd = rsqrtf(var + 1e-5f);

  bf16x8 o;
#pragma unroll
  for (int j = 0; j < 8; ++j) {
    int c = t * 8 + j;
    float xv = (j < 4) ? v0[j] : v1[j - 4];
    o[j] = (__bf16)((xv - mu) * rstd * w[c] + bta[c]);
  }
  *(AS1 bf16x8*)((AS1 __bf16*)xn + (size_t)row * Hh + t * 8) = o;
}

// ---------------------------------------------------------------------------
// Shared GEMM mainloop: C[128x128] tile, BK=64, double-buffered LDS fed by
// async global->LDS B128 DMA. A: [M][K] bf16. Bt: [N][K] bf16 (pre-
// transposed). 8 waves; each computes 32x64 via 2x4 WMMA tiles.
// ---------------------------------------------------------------------------
static __device__ __forceinline__ void gemm_stage(const AS1 __bf16* gA,
                                                  const AS1 __bf16* gB,
                                                  AS3 __bf16* sA, AS3 __bf16* sB,
                                                  int t) {
#pragma unroll
  for (int i = 0; i < 4; ++i) {
    int idx = i * 256 + t;            // 1024 octets per 128x64 tile
    int row = idx >> 3;
    int c8  = (idx & 7) << 3;
    cp16_g2l(gA + (size_t)row * Hh + c8, sA + row * 64 + c8);
    cp16_g2l(gB + (size_t)row * Hh + c8, sB + row * 64 + c8);
  }
}

static __device__ __forceinline__ void gemm_mainloop(const AS1 __bf16* gA,
                                                     const AS1 __bf16* gB,
                                                     AS3 __bf16* As0, AS3 __bf16* As1,
                                                     AS3 __bf16* Bs0, AS3 __bf16* Bs1,
                                                     int wm, int wn, int t,
                                                     f32x8 acc[2][4]) {
  const int ntile = Hh / 64;
  gemm_stage(gA, gB, As0, Bs0, t);
  for (int kb = 0; kb < ntile; ++kb) {
    const int cur = kb & 1;
    async_wait0();                 // this wave's DMA for tile kb done
    __syncthreads();               // all waves staged; prior readers done
    if (kb + 1 < ntile)
      gemm_stage(gA + (kb + 1) * 64, gB + (kb + 1) * 64,
                 cur ? As0 : As1, cur ? Bs0 : Bs1, t);
    const AS3 __bf16* sa = cur ? As1 : As0;
    const AS3 __bf16* sb = cur ? Bs1 : Bs0;
#pragma unroll
    for (int kk = 0; kk < 64; kk += 32) {
      bf16x16 a0 = load_frag_l(sa + (wm * 32 +  0) * 64 + kk, 64);
      bf16x16 a1 = load_frag_l(sa + (wm * 32 + 16) * 64 + kk, 64);
      bf16x16 b0 = load_frag_l(sb + (wn * 64 +  0) * 64 + kk, 64);
      bf16x16 b1 = load_frag_l(sb + (wn * 64 + 16) * 64 + kk, 64);
      bf16x16 b2 = load_frag_l(sb + (wn * 64 + 32) * 64 + kk, 64);
      bf16x16 b3 = load_frag_l(sb + (wn * 64 + 48) * 64 + kk, 64);
      acc[0][0] = wmma_bf16(a0, b0, acc[0][0]);
      acc[0][1] = wmma_bf16(a0, b1, acc[0][1]);
      acc[0][2] = wmma_bf16(a0, b2, acc[0][2]);
      acc[0][3] = wmma_bf16(a0, b3, acc[0][3]);
      acc[1][0] = wmma_bf16(a1, b0, acc[1][0]);
      acc[1][1] = wmma_bf16(a1, b1, acc[1][1]);
      acc[1][2] = wmma_bf16(a1, b2, acc[1][2]);
      acc[1][3] = wmma_bf16(a1, b3, acc[1][3]);
    }
    __syncthreads();               // readers done before next overwrite
  }
}

// ---------------------------------------------------------------------------
// QKV GEMM: [8192,2048] x Bt[6144,2048] + bias -> bf16 Q/K [B,NH,S,HD],
// V transposed [B,NH,HD,S] (free in the scatter epilogue).
// ---------------------------------------------------------------------------
__global__ __launch_bounds__(256) void qkv_gemm(const __bf16* __restrict__ A,
                                                const __bf16* __restrict__ Bt,
                                                const float* __restrict__ bias,
                                                __bf16* __restrict__ qd,
                                                __bf16* __restrict__ kd,
                                                __bf16* __restrict__ vtd) {
  __shared__ __align__(16) __bf16 As[2][128 * 64];
  __shared__ __align__(16) __bf16 Bs[2][128 * 64];
  const int t = threadIdx.x;
  const int lane = t & 31, wv = t >> 5;
  const int wm = wv >> 1, wn = wv & 1;
  const int nBase = blockIdx.x * 128;
  const int mBase = blockIdx.y * 128;

  f32x8 acc[2][4];
#pragma unroll
  for (int i = 0; i < 2; ++i)
#pragma unroll
    for (int j = 0; j < 4; ++j) acc[i][j] = f32x8_zero();

  gemm_mainloop((const AS1 __bf16*)A  + (size_t)mBase * Hh,
                (const AS1 __bf16*)Bt + (size_t)nBase * Hh,
                (AS3 __bf16*)As[0], (AS3 __bf16*)As[1],
                (AS3 __bf16*)Bs[0], (AS3 __bf16*)Bs[1], wm, wn, t, acc);

  const AS1 float* gb  = (const AS1 float*)bias;
  AS1 __bf16* gq = (AS1 __bf16*)qd;
  AS1 __bf16* gk = (AS1 __bf16*)kd;
  AS1 __bf16* gv = (AS1 __bf16*)vtd;
  const int halfOff = (lane >> 4) << 3;
  const int nc = lane & 15;
#pragma unroll
  for (int mi = 0; mi < 2; ++mi)
#pragma unroll
    for (int ni = 0; ni < 4; ++ni)
#pragma unroll
      for (int r = 0; r < 8; ++r) {
        int m = mBase + wm * 32 + mi * 16 + halfOff + r;
        int n = nBase + wn * 64 + ni * 16 + nc;
        float v = acc[mi][ni][r] + gb[n];
        int b = m >> 11, sidx = m & (Ss - 1);
        int sel = n >> 11, hp = n & (Hh - 1);
        int nh = hp >> 7, d = hp & (HDd - 1);
        __bf16 bv = (__bf16)v;
        if (sel == 0)
          gq[(((size_t)(b * NHh + nh)) * Ss + sidx) * HDd + d] = bv;
        else if (sel == 1)
          gk[(((size_t)(b * NHh + nh)) * Ss + sidx) * HDd + d] = bv;
        else
          gv[(((size_t)(b * NHh + nh)) * HDd + d) * Ss + sidx] = bv;
      }
}

// ---------------------------------------------------------------------------
// Causal flash attention. One (b, nh, 128-q-row tile) per WG; 8 waves of 16
// q rows. KV blocks of 64. Q/K frags direct from global, V tile DMA'd async
// from pre-transposed vt while the wave computes QK^T + softmax.
// ---------------------------------------------------------------------------
__global__ __launch_bounds__(256) void flash_attn(const __bf16* __restrict__ q,
                                                  const __bf16* __restrict__ k,
                                                  const __bf16* __restrict__ vt,
                                                  __bf16* __restrict__ o) {
  __shared__ __align__(16) __bf16 VtS[128 * 64];      // [d][kv]
  __shared__ __align__(16) __bf16 PwS[8 * 16 * 64];   // per-wave P
  const int t = threadIdx.x;
  const int lane = t & 31, wv = t >> 5;
  const int qb = blockIdx.x & 15;
  const int nh = (blockIdx.x >> 4) & 15;
  const int b  = blockIdx.x >> 8;
  const size_t headOff = ((size_t)(b * NHh + nh)) * Ss * HDd;
  const AS1 __bf16* qp  = (const AS1 __bf16*)q  + headOff;
  const AS1 __bf16* kp  = (const AS1 __bf16*)k  + headOff;
  const AS1 __bf16* vtp = (const AS1 __bf16*)vt + headOff;   // [HD][S]
  AS3 __bf16* Vt = (AS3 __bf16*)VtS;
  AS3 __bf16* Pwv = (AS3 __bf16*)PwS + wv * 16 * 64;
  const int qBase = qb * 128;
  const int qRow0 = qBase + wv * 16;
  const int halfOff = (lane >> 4) << 3;
  const int nc = lane & 15;
  const float scale = 0.08838834764831845f;   // 1/sqrt(128)

  bf16x16 qf[4];
#pragma unroll
  for (int f = 0; f < 4; ++f)
    qf[f] = load_frag_g(qp + (size_t)qRow0 * HDd + f * 32, HDd);

  f32x8 oacc[8];
#pragma unroll
  for (int dt = 0; dt < 8; ++dt) oacc[dt] = f32x8_zero();
  float mrun[8], lrun[8];
#pragma unroll
  for (int r = 0; r < 8; ++r) { mrun[r] = -__builtin_inff(); lrun[r] = 0.f; }

  const int nkv = qb * 2 + 2;

  for (int kvb = 0; kvb < nkv; ++kvb) {
    const int kvBase = kvb * 64;
    __syncthreads();                       // prior P*V readers done with Vt
    // issue V-tile DMA: Vt[d][kv] <- vt[d][kvBase+kv], fully contiguous
#pragma unroll
    for (int i = 0; i < 4; ++i) {
      int idx = i * 256 + t;               // 1024 octets (128 x 64)
      int d  = idx >> 3;
      int c8 = (idx & 7) << 3;
      cp16_g2l(vtp + (size_t)d * Ss + kvBase + c8, Vt + d * 64 + c8);
    }

    // scores S = Q K^T while the DMA is in flight
    f32x8 sacc[4];
#pragma unroll
    for (int nt = 0; nt < 4; ++nt) {
      sacc[nt] = f32x8_zero();
#pragma unroll
      for (int f = 0; f < 4; ++f) {
        bf16x16 kf = load_frag_g(kp + (size_t)(kvBase + nt * 16) * HDd + f * 32, HDd);
        sacc[nt] = wmma_bf16(qf[f], kf, sacc[nt]);
      }
    }

    // scale + causal mask + row max
    float rmax[8];
#pragma unroll
    for (int r = 0; r < 8; ++r) rmax[r] = -1e30f;
#pragma unroll
    for (int nt = 0; nt < 4; ++nt)
#pragma unroll
      for (int r = 0; r < 8; ++r) {
        float sv = sacc[nt][r] * scale;
        int qAbs = qRow0 + halfOff + r;
        int kAbs = kvBase + nt * 16 + nc;
        sv = (kAbs <= qAbs) ? sv : -1e30f;
        sacc[nt][r] = sv;
        rmax[r] = fmaxf(rmax[r], sv);
      }
#pragma unroll
    for (int r = 0; r < 8; ++r)
#pragma unroll
      for (int m = 1; m <= 8; m <<= 1)
        rmax[r] = fmaxf(rmax[r], __shfl_xor(rmax[r], m, 32));

    // online softmax update
    float rsum[8];
#pragma unroll
    for (int r = 0; r < 8; ++r) {
      float mn = fmaxf(mrun[r], rmax[r]);
      float corr = __expf(mrun[r] - mn);
      mrun[r] = mn;
      lrun[r] *= corr;
#pragma unroll
      for (int dt = 0; dt < 8; ++dt) oacc[dt][r] *= corr;
      float rs = 0.f;
#pragma unroll
      for (int nt = 0; nt < 4; ++nt) {
        float p = __expf(sacc[nt][r] - mn);
        sacc[nt][r] = p;
        rs += p;
      }
      rsum[r] = rs;
    }
#pragma unroll
    for (int r = 0; r < 8; ++r) {
#pragma unroll
      for (int m = 1; m <= 8; m <<= 1)
        rsum[r] += __shfl_xor(rsum[r], m, 32);
      lrun[r] += rsum[r];
    }

    // write P (bf16) to per-wave LDS (C-layout -> row-major)
#pragma unroll
    for (int nt = 0; nt < 4; ++nt)
#pragma unroll
      for (int r = 0; r < 8; ++r)
        Pwv[(halfOff + r) * 64 + nt * 16 + nc] = (__bf16)sacc[nt][r];

    async_wait0();                         // V tile landed
    __syncthreads();

    // O += P V
#pragma unroll
    for (int f = 0; f < 2; ++f) {
      bf16x16 pf = load_frag_l(Pwv + f * 32, 64);
#pragma unroll
      for (int dt = 0; dt < 8; ++dt) {
        bf16x16 vf = load_frag_l(Vt + (dt * 16) * 64 + f * 32, 64);
        oacc[dt] = wmma_bf16(pf, vf, oacc[dt]);
      }
    }
  }

  // normalize and merge heads -> o[B,S,H] (bf16)
  AS1 __bf16* go = (AS1 __bf16*)o;
  float inv[8];
#pragma unroll
  for (int r = 0; r < 8; ++r) inv[r] = 1.0f / lrun[r];
#pragma unroll
  for (int dt = 0; dt < 8; ++dt)
#pragma unroll
    for (int r = 0; r < 8; ++r) {
      int srow = qRow0 + halfOff + r;
      int col = nh * HDd + dt * 16 + nc;
      go[((size_t)(b * Ss + srow)) * Hh + col] = (__bf16)(oacc[dt][r] * inv[r]);
    }
}

// ---------------------------------------------------------------------------
// Output projection: [8192,2048] x Bt[2048,2048] + bias + residual -> fp32
// ---------------------------------------------------------------------------
__global__ __launch_bounds__(256) void proj_gemm(const __bf16* __restrict__ A,
                                                 const __bf16* __restrict__ Bt,
                                                 const float* __restrict__ bias,
                                                 const float* __restrict__ resid,
                                                 float* __restrict__ out) {
  __shared__ __align__(16) __bf16 As[2][128 * 64];
  __shared__ __align__(16) __bf16 Bs[2][128 * 64];
  const int t = threadIdx.x;
  const int lane = t & 31, wv = t >> 5;
  const int wm = wv >> 1, wn = wv & 1;
  const int nBase = blockIdx.x * 128;
  const int mBase = blockIdx.y * 128;

  f32x8 acc[2][4];
#pragma unroll
  for (int i = 0; i < 2; ++i)
#pragma unroll
    for (int j = 0; j < 4; ++j) acc[i][j] = f32x8_zero();

  gemm_mainloop((const AS1 __bf16*)A  + (size_t)mBase * Hh,
                (const AS1 __bf16*)Bt + (size_t)nBase * Hh,
                (AS3 __bf16*)As[0], (AS3 __bf16*)As[1],
                (AS3 __bf16*)Bs[0], (AS3 __bf16*)Bs[1], wm, wn, t, acc);

  const AS1 float* gb = (const AS1 float*)bias;
  const AS1 float* gr = (const AS1 float*)resid;
  AS1 float* go = (AS1 float*)out;
  const int halfOff = (lane >> 4) << 3;
  const int nc = lane & 15;
#pragma unroll
  for (int mi = 0; mi < 2; ++mi)
#pragma unroll
    for (int ni = 0; ni < 4; ++ni)
#pragma unroll
      for (int r = 0; r < 8; ++r) {
        int m = mBase + wm * 32 + mi * 16 + halfOff + r;
        int n = nBase + wn * 64 + ni * 16 + nc;
        size_t off = (size_t)m * Hh + n;
        go[off] = acc[mi][ni][r] + gb[n] + gr[off];
      }
}

// ---------------------------------------------------------------------------
// Host-side launcher
// ---------------------------------------------------------------------------
extern "C" void kernel_launch(void* const* d_in, const int* in_sizes, int n_in,
                              void* d_out, int out_size, void* d_ws, size_t ws_size,
                              hipStream_t stream) {
  (void)in_sizes; (void)n_in; (void)out_size; (void)ws_size;
  const float* x       = (const float*)d_in[0];
  const float* ln_w    = (const float*)d_in[1];
  const float* ln_b    = (const float*)d_in[2];
  const float* attn_w  = (const float*)d_in[3];
  const float* attn_b  = (const float*)d_in[4];
  const float* proj_w  = (const float*)d_in[5];
  const float* proj_b  = (const float*)d_in[6];
  float* out = (float*)d_out;

  char* ws = (char*)d_ws;
  const size_t TOK = (size_t)Bb * Ss;                 // 8192
  const size_t SZ_ACT = TOK * Hh * sizeof(__bf16);    // 32 MiB
  __bf16* xn    = (__bf16*)(ws);
  __bf16* qws   = (__bf16*)(ws + 1 * SZ_ACT);
  __bf16* kws   = (__bf16*)(ws + 2 * SZ_ACT);
  __bf16* vtws  = (__bf16*)(ws + 3 * SZ_ACT);         // [B,NH,HD,S]
  __bf16* ows   = (__bf16*)(ws + 4 * SZ_ACT);
  __bf16* wqkvT = (__bf16*)(ws + 5 * SZ_ACT);                         // [6144][2048], 24 MiB
  __bf16* wprjT = (__bf16*)(ws + 5 * SZ_ACT + (size_t)Hh * N3 * 2);   // [2048][2048],  8 MiB

  // one-time weight transpose + bf16 convert: [K][N]f32 -> [N][K]bf16
  cvt_transpose_bf16<<<dim3(N3 / 64, Hh / 64), 256, 0, stream>>>(attn_w, wqkvT, Hh, N3);
  cvt_transpose_bf16<<<dim3(Hh / 64, Hh / 64), 256, 0, stream>>>(proj_w, wprjT, Hh, Hh);

  // fused LayerNorm + bf16 cast
  ln_bf16<<<TOK, 256, 0, stream>>>(x, ln_w, ln_b, xn);

  // QKV projection
  qkv_gemm<<<dim3(N3 / 128, TOK / 128), 256, 0, stream>>>(xn, wqkvT, attn_b,
                                                          qws, kws, vtws);

  // causal flash attention
  flash_attn<<<Bb * NHh * (Ss / 128), 256, 0, stream>>>(qws, kws, vtws, ows);

  // output projection + bias + residual
  proj_gemm<<<dim3(Hh / 128, TOK / 128), 256, 0, stream>>>(ows, wprjT, proj_b,
                                                           x, out);
}
